// Wavenet_26946624815210
// MI455X (gfx1250) — matrix operand order repaired and checked by reading, not compiled, not run on hardware
//
#include <hip/hip_runtime.h>

// ---------------------------------------------------------------------------
// WaveNet forward for MI455X (gfx1250, wave32, WMMA f32_16x16x32_f16).
//
// Coordinates: everything indexed by offset-from-the-end e (reference always
// trims from the front). Layer recurrence in end coords:
//   z[:,e]  = W0 @ h[:,e+d+1] + W1 @ h[:,e+1]
//   g       = tanh(z)*sigmoid(z) = (1-exp(-z)) / (1+exp(-2z))
//   h'[:,e] = Wres @ g[:,e] + b_res + h[:,e]
//   skip_sum[:,e] = sum_i (Wskip_i @ g_i[:,e]) + sum_i b_skip_i   (e < final_len)
//   out = Wd2 @ relu(Wd1 @ relu(skip_sum)), written time-reversed (j = FL-1-e).
//
// Activations f16 stored [time][64ch] -> B fragments are one contiguous 32B
// load/lane; weights f16 row-major [co][ci] -> A fragments are two 16B
// loads/lane; A and B share one contiguous k order (k = 16*hi + element).
//
// wn_skipfinal uses N=32 per wave (two B column tiles share each A fragment)
// to halve the L2 re-stream of the 640KB skip-weight matrix; gfx1250's
// 1024-VGPR/wave budget covers the 2x accumulator set.
// ---------------------------------------------------------------------------

typedef _Float16 half_t;
typedef half_t v16h __attribute__((ext_vector_type(16)));
typedef half_t v8h  __attribute__((ext_vector_type(8)));
typedef float  v8f  __attribute__((ext_vector_type(8)));

#define NLAYER 20
#define RESC   64
#define SKIPC  256

#define WMMA(A, B, C) \
  __builtin_amdgcn_wmma_f32_16x16x32_f16(false, (A), false, (B), (short)0, (C), false, false)

// B fragment (K x 16, f16): lane holds column n=lane&15; element p -> K = p + 16*hi.
static __device__ __forceinline__ v16h ldB(const half_t* __restrict__ row, int ks, int hi) {
  return *(const v16h*)(row + ks * 32 + hi * 16);
}
// A fragment (16 x K, f16): lane holds row m=lane&15; element p -> K = p + 16*hi.
static __device__ __forceinline__ v16h ldA(const half_t* __restrict__ wrow, int ks, int hi) {
  v16h a;
  *((v8h*)&a)     = *(const v8h*)(wrow + ks * 32 + hi * 16);
  *((v8h*)&a + 1) = *(const v8h*)(wrow + ks * 32 + hi * 16 + 8);
  return a;
}

// tanh(z)*sigmoid(z) == (1-u)/(1+u*u), u = exp(-z). One exp + one v_rcp_f32.
static __device__ __forceinline__ float gatefn(float z) {
  float u = __expf(-z);
  return (1.0f - u) * __builtin_amdgcn_rcpf(1.0f + u * u);
}

// ---------------------------------------------------------------------------
// Weight conversion / relayout to f16.
// WDIL  [i][tap][64co][64ci]   from w_dil  (i,co,ci,tap)
// WRES  [i][64co][64ci]        from w_res  (identical flat order, K=1 dropped)
// WSKIP [i][256co][64ci]       from w_skip (identical flat order)
// WD1/WD2 [256co][256ci]       straight convert
// WCT   [tok][tap][64c]        from w_causal (c,tok,tap)   (one-hot gather table)
// ---------------------------------------------------------------------------
__global__ void __launch_bounds__(256) wn_convert(
    const float* __restrict__ w_causal, const float* __restrict__ w_dil,
    const float* __restrict__ w_res, const float* __restrict__ w_skip,
    const float* __restrict__ w_d1, const float* __restrict__ w_d2,
    half_t* __restrict__ WCT, half_t* __restrict__ WDIL,
    half_t* __restrict__ WRES, half_t* __restrict__ WSKIP,
    half_t* __restrict__ WD1, half_t* __restrict__ WD2) {
  int idx = blockIdx.x * 256 + threadIdx.x;
  if (idx < 163840) {  // WDIL: tap de-interleave
    int i = idx >> 13, r = idx & 8191;
    int tap = r >> 12, co = (r >> 6) & 63, ci = r & 63;
    WDIL[idx] = (half_t)w_dil[(((i * 64 + co) * 64 + ci) << 1) + tap];
    return;
  }
  idx -= 163840;
  if (idx < 81920)  { WRES[idx]  = (half_t)w_res[idx];  return; }  idx -= 81920;
  if (idx < 327680) { WSKIP[idx] = (half_t)w_skip[idx]; return; }  idx -= 327680;
  if (idx < 65536)  { WD1[idx]   = (half_t)w_d1[idx];   return; }  idx -= 65536;
  if (idx < 65536)  { WD2[idx]   = (half_t)w_d2[idx];   return; }  idx -= 65536;
  if (idx < 32768) {  // WCT: transpose (c,tok,tap) -> (tok,tap,c)
    int tok = idx >> 7, r = idx & 127, tap = r >> 6, c = r & 63;
    WCT[idx] = (half_t)w_causal[((c * 256 + tok) << 1) + tap];
  }
}

__global__ void wn_sbias(const float* __restrict__ b_skip, float* __restrict__ sb) {
  int c = threadIdx.x;  // 256 threads
  float s = 0.f;
  for (int i = 0; i < NLAYER; ++i) s += b_skip[i * 256 + c];
  sb[c] = s;
}

// Initial causal conv on one-hot input = gather of two WCT rows.
// h0[c, j] = WCT[tok(j-1)][0][c] + WCT[tok(j)][1][c], stored at e = T-1-j.
__global__ void __launch_bounds__(256) wn_init(const int* __restrict__ x,
                                               const half_t* __restrict__ WCT,
                                               half_t* __restrict__ h0, int T) {
  const int idx = blockIdx.x * 256 + threadIdx.x;  // T*4 threads
  const int j = idx >> 2, q = idx & 3;
  if (j >= T) return;
  const int a = (j == 0) ? 0 : x[j - 1];  // front zero-pad one-hots token 0
  const int b = x[j];
  v16h va = *(const v16h*)(WCT + a * 128 + q * 16);
  v16h vb = *(const v16h*)(WCT + b * 128 + 64 + q * 16);
  v16h o;
#pragma unroll
  for (int r = 0; r < 16; ++r) o[r] = (half_t)((float)va[r] + (float)vb[r]);
  *(v16h*)(h0 + (size_t)(T - 1 - j) * RESC + q * 16) = o;
}

// ---------------------------------------------------------------------------
// One residual block. 8 waves/block, 16 time steps/wave.
// Per wave-tile: 16 WMMAs (dilated conv), gate, 8 WMMAs (res 1x1), residual.
// Weights (24KB/layer) live in WGP$, so A reloads are on-chip; kernel is
// bound by the h/g HBM streams. g staged via per-wave LDS for the
// C/D-layout -> B-layout transpose, and written to global for the skip GEMM.
// ---------------------------------------------------------------------------
__global__ void __launch_bounds__(256) wn_layer(
    const half_t* __restrict__ hin, half_t* __restrict__ hout,
    half_t* __restrict__ gout,
    const half_t* __restrict__ W0, const half_t* __restrict__ W1,
    const half_t* __restrict__ WR, const float* __restrict__ brs, int d) {
  __shared__ __align__(32) half_t gls[8][16][RESC];
  const int wave = threadIdx.x >> 5;
  const int lane = threadIdx.x & 31;
  const int n = lane & 15, hi = lane >> 4;
  const int e = (blockIdx.x * 8 + wave) * 16 + n;

  const half_t* rowF = hin + (size_t)(e + d + 1) * RESC;  // tap 0 (older sample)
  const half_t* rowN = hin + (size_t)(e + 1) * RESC;      // tap 1 (newer sample)
  __builtin_prefetch(hin + (size_t)(e + d + 1 + 128) * RESC, 0, 1);  // next block's far tap

  v8f acc[4] = {};
#pragma unroll
  for (int ks = 0; ks < 2; ++ks) {
    v16h b0 = ldB(rowF, ks, hi);
    v16h b1 = ldB(rowN, ks, hi);
#pragma unroll
    for (int mt = 0; mt < 4; ++mt) {
      acc[mt] = WMMA(ldA(W0 + (size_t)(mt * 16 + n) * RESC, ks, hi), b0, acc[mt]);
      acc[mt] = WMMA(ldA(W1 + (size_t)(mt * 16 + n) * RESC, ks, hi), b1, acc[mt]);
    }
  }

  // gated activation -> f16; stage to LDS (res-GEMM B) + global (skip GEMM)
#pragma unroll
  for (int mt = 0; mt < 4; ++mt) {
    v8h gv;
#pragma unroll
    for (int r = 0; r < 8; ++r) gv[r] = (half_t)gatefn(acc[mt][r]);
    const int cb = mt * 16 + hi * 8;  // D-layout channel: r + 8*hi + 16*mt
    *(v8h*)&gls[wave][n][cb] = gv;
    *(v8h*)(gout + (size_t)e * RESC + cb) = gv;
  }

  v8f accR[4] = {};
#pragma unroll
  for (int ks = 0; ks < 2; ++ks) {
    v16h bg = *(const v16h*)&gls[wave][n][ks * 32 + hi * 16];
#pragma unroll
    for (int mt = 0; mt < 4; ++mt)
      accR[mt] = WMMA(ldA(WR + (size_t)(mt * 16 + n) * RESC, ks, hi), bg, accR[mt]);
  }

#pragma unroll
  for (int mt = 0; mt < 4; ++mt) {
    const int cb = mt * 16 + hi * 8;
    v8h ho = *(const v8h*)(hin + (size_t)e * RESC + cb);
    v8h hv;
#pragma unroll
    for (int r = 0; r < 8; ++r)
      hv[r] = (half_t)(accR[mt][r] + brs[cb + r] + (float)ho[r]);
    *(v8h*)(hout + (size_t)e * RESC + cb) = hv;
  }
}

// ---------------------------------------------------------------------------
// Fused skip-sum + head: skip = sum_i Wskip_i @ g_i (K = 20*64 = 1280),
// + summed bias, ReLU, Wd1 (K=256), ReLU, Wd2 (K=256), store f32 reversed.
// 4 waves/block, N=32 per wave for the skip phase: each 640KB-weight A
// fragment feeds two column tiles (halves L2 weight traffic). Head GEMMs run
// per 16-column half through an 8KB/wave LDS staging buffer (32KB/block).
// ---------------------------------------------------------------------------
__global__ void __launch_bounds__(128) wn_skipfinal(
    const half_t* __restrict__ G, unsigned long long gstride,
    const half_t* __restrict__ WSK, const float* __restrict__ sbias,
    const half_t* __restrict__ WD1, const half_t* __restrict__ WD2,
    float* __restrict__ out, int final_len) {
  __shared__ __align__(32) half_t yls[4][16][SKIPC];
  const int wave = threadIdx.x >> 5;
  const int lane = threadIdx.x & 31;
  const int n = lane & 15, hi = lane >> 4;
  const int t0 = (blockIdx.x * 4 + wave) * 32;  // 32 time columns per wave
  const int e0 = t0 + n;
  const int e1 = t0 + 16 + n;

  // skip accumulation over all 20 layers, two column tiles per A fragment
  v8f acc[16][2] = {};
  for (int i = 0; i < NLAYER; ++i) {
    const half_t* g0 = G + (size_t)i * gstride + (size_t)e0 * RESC;
    const half_t* g1 = G + (size_t)i * gstride + (size_t)e1 * RESC;
    const half_t* Wl = WSK + (size_t)i * (SKIPC * RESC);
#pragma unroll
    for (int ks = 0; ks < 2; ++ks) {
      v16h b0 = *(const v16h*)(g0 + ks * 32 + hi * 16);
      v16h b1 = *(const v16h*)(g1 + ks * 32 + hi * 16);
#pragma unroll
      for (int mt = 0; mt < 16; ++mt) {
        v16h a = ldA(Wl + (size_t)(mt * 16 + n) * RESC, ks, hi);
        acc[mt][0] = WMMA(a, b0, acc[mt][0]);
        acc[mt][1] = WMMA(a, b1, acc[mt][1]);
      }
    }
  }

  // head: process each 16-column half through the per-wave LDS buffer
  for (int nt = 0; nt < 2; ++nt) {
    const int e = t0 + nt * 16 + n;

    // skip bias + ReLU -> f16 staging (D-layout -> B-layout via LDS)
#pragma unroll
    for (int mt = 0; mt < 16; ++mt) {
      const int cb = mt * 16 + hi * 8;
      v8h yv;
#pragma unroll
      for (int r = 0; r < 8; ++r) {
        float v = acc[mt][nt][r] + sbias[cb + r];
        yv[r] = (half_t)(v > 0.f ? v : 0.f);
      }
      *(v8h*)&yls[wave][n][cb] = yv;
    }

    v8f a1[16] = {};
#pragma unroll
    for (int ks = 0; ks < 8; ++ks) {
      v16h b = *(const v16h*)&yls[wave][n][ks * 32 + hi * 16];
#pragma unroll
      for (int mt = 0; mt < 16; ++mt)
        a1[mt] = WMMA(ldA(WD1 + (size_t)(mt * 16 + n) * SKIPC, ks, hi), b, a1[mt]);
    }
#pragma unroll
    for (int mt = 0; mt < 16; ++mt) {
      const int cb = mt * 16 + hi * 8;
      v8h yv;
#pragma unroll
      for (int r = 0; r < 8; ++r) {
        float v = a1[mt][r];
        yv[r] = (half_t)(v > 0.f ? v : 0.f);
      }
      *(v8h*)&yls[wave][n][cb] = yv;  // same-wave LDS reuse: all reads precede writes
    }

    v8f a2[16] = {};
#pragma unroll
    for (int ks = 0; ks < 8; ++ks) {
      v16h b = *(const v16h*)&yls[wave][n][ks * 32 + hi * 16];
#pragma unroll
      for (int mt = 0; mt < 16; ++mt)
        a2[mt] = WMMA(ldA(WD2 + (size_t)(mt * 16 + n) * SKIPC, ks, hi), b, a2[mt]);
    }

    if (e < final_len) {  // divergence only after the last WMMA of this half
      const int j = final_len - 1 - e;  // un-reverse time
#pragma unroll
      for (int mt = 0; mt < 16; ++mt) {
        const int cb = mt * 16 + hi * 8;
#pragma unroll
        for (int r = 0; r < 8; ++r)
          out[(size_t)(cb + r) * final_len + j] = a2[mt][r];
      }
    }
  }
}

// ---------------------------------------------------------------------------
extern "C" void kernel_launch(void* const* d_in, const int* in_sizes, int n_in,
                              void* d_out, int out_size, void* d_ws, size_t ws_size,
                              hipStream_t stream) {
  (void)n_in; (void)ws_size;  // needs ~190 MB workspace
  const int T = in_sizes[0];
  const int* x          = (const int*)d_in[0];
  const float* w_causal = (const float*)d_in[1];
  const float* w_dil    = (const float*)d_in[2];
  const float* w_res    = (const float*)d_in[3];
  const float* b_res    = (const float*)d_in[4];
  const float* w_skip   = (const float*)d_in[5];
  const float* b_skip   = (const float*)d_in[6];
  const float* w_d1     = (const float*)d_in[7];
  const float* w_d2     = (const float*)d_in[8];
  float* out = (float*)d_out;
  const int final_len = out_size / 256;

  const int EN = T + 1024;                   // end-aligned rows + tile/dilation slack
  const size_t hrow = (size_t)EN * RESC;     // halves per activation buffer

  char* p = (char*)d_ws;
  auto take = [&](size_t bytes) -> char* {
    char* r = p; p += (bytes + 255) & ~(size_t)255; return r;
  };
  half_t* hA    = (half_t*)take(hrow * 2);
  half_t* hB    = (half_t*)take(hrow * 2);
  half_t* G     = (half_t*)take(hrow * 2 * NLAYER);
  half_t* WCT   = (half_t*)take((size_t)256 * 2 * 64 * 2);
  half_t* WDIL  = (half_t*)take((size_t)NLAYER * 2 * 64 * 64 * 2);
  half_t* WRES  = (half_t*)take((size_t)NLAYER * 64 * 64 * 2);
  half_t* WSKIP = (half_t*)take((size_t)NLAYER * 256 * 64 * 2);
  half_t* WD1   = (half_t*)take((size_t)256 * 256 * 2);
  half_t* WD2   = (half_t*)take((size_t)256 * 256 * 2);
  float*  SBS   = (float*)take(256 * 4);

  wn_convert<<<(737280 + 255) / 256, 256, 0, stream>>>(
      w_causal, w_dil, w_res, w_skip, w_d1, w_d2, WCT, WDIL, WRES, WSKIP, WD1, WD2);
  wn_sbias<<<1, 256, 0, stream>>>(b_skip, SBS);
  wn_init<<<(T * 4 + 255) / 256, 256, 0, stream>>>(x, WCT, hA, T);

  static const int DILS[NLAYER] = {1, 2, 4, 8, 16, 32, 64, 128, 256, 512,
                                   1, 2, 4, 8, 16, 32, 64, 128, 256, 512};
  half_t* hin = hA;
  half_t* hout = hB;
  int L = T;
  for (int i = 0; i < NLAYER; ++i) {
    const int d = DILS[i];
    const int Lnew = L - d - 1;
    const int blocks = (Lnew + 127) / 128;
    wn_layer<<<blocks, 256, 0, stream>>>(hin, hout, G + (size_t)i * hrow,
                                         WDIL + (size_t)(2 * i) * 4096,
                                         WDIL + (size_t)(2 * i + 1) * 4096,
                                         WRES + (size_t)i * 4096,
                                         b_res + (size_t)i * 64, d);
    half_t* t = hin; hin = hout; hout = t;
    L = Lnew;
  }

  // 4 waves x 32 cols = 128 columns per block
  wn_skipfinal<<<(final_len + 127) / 128, 128, 0, stream>>>(
      G, (unsigned long long)hrow, WSKIP, SBS, WD1, WD2, out, final_len);
}
